// WN_52286931861792
// MI455X (gfx1250) — matrix operand order repaired and checked by reading, not compiled, not run on hardware
//
#include <hip/hip_runtime.h>
#include <hip/hip_bf16.h>
#include <string.h>

// ---------------------------------------------------------------------------
// WaveNet on gfx1250: all channel-space convs as bf16 WMMA GEMMs (f32 accum).
// ---------------------------------------------------------------------------

#define BATCH 32
#define TLEN  4000
#define NC    128
#define KS    3
#define N_IN  4
#define N_EMB 80
#define NLAY  7

#define TT    64            // time tile per workgroup
#define KTOT  480           // padded concat-K for GEMM1 (464 real + 16 zero)
#define KPAD  488           // LDS row stride (elements) for feat tile

typedef __attribute__((ext_vector_type(16))) __bf16 v16bf;
typedef __attribute__((ext_vector_type(8)))  float  v8f;

union Frag16 { v16bf v; uint4 q[2]; };

__device__ __forceinline__ unsigned short f2bf(float f) {
    unsigned int u = __float_as_uint(f);
    u += 0x7FFFu + ((u >> 16) & 1u);        // round-to-nearest-even
    return (unsigned short)(u >> 16);
}

// ---------------------------------------------------------------------------
// Weight / bias packing:  Wcat[l][m][k] bf16 (k: [x(t-d)|x(t)|x(t+d)|mel|0]),
//                         Wrs[l][m][c] bf16, biasCat/biasRs f32.
// ---------------------------------------------------------------------------
__global__ void prep_weights(const float* __restrict__ in_w,
                             const float* __restrict__ in_b,
                             const float* __restrict__ cond_w,
                             const float* __restrict__ cond_b,
                             const float* __restrict__ rs_w,
                             const float* __restrict__ rs_b,
                             const float* __restrict__ rs_w_last,
                             const float* __restrict__ rs_b_last,
                             unsigned short* __restrict__ Wcat,
                             unsigned short* __restrict__ Wrs,
                             float* __restrict__ biasCat,
                             float* __restrict__ biasRs)
{
    const int NW1 = NLAY * 256 * KTOT;
    const int NW2 = NLAY * 256 * NC;
    const int NB  = NLAY * 256;
    int idx = blockIdx.x * blockDim.x + threadIdx.x;

    if (idx < NW1) {                                  // Wcat
        int l = idx / (256 * KTOT);
        int r = idx % (256 * KTOT);
        int m = r / KTOT;
        int k = r % KTOT;
        float v = 0.f;
        if (k < 3 * NC) {
            int j = k / NC, c = k % NC;               // tap j: 0->t-d,1->t,2->t+d
            v = in_w[(((size_t)l * 256 + m) * NC + c) * KS + j];
        } else if (k < 3 * NC + N_EMB) {
            int e = k - 3 * NC;
            v = cond_w[((size_t)l * 256 + m) * N_EMB + e];
        }
        Wcat[idx] = f2bf(v);
        return;
    }
    idx -= NW1;
    if (idx < NW2) {                                  // Wrs
        int l = idx / (256 * NC);
        int m = (idx / NC) % 256;
        int c = idx % NC;
        float v = 0.f;
        if (l < NLAY - 1)            v = rs_w[((size_t)l * 256 + m) * NC + c];
        else if (m < NC)             v = rs_w_last[(size_t)m * NC + c];
        Wrs[idx] = f2bf(v);
        return;
    }
    idx -= NW2;
    if (idx < NB) {                                   // biasCat
        biasCat[idx] = in_b[idx] + cond_b[idx];
        return;
    }
    idx -= NB;
    if (idx < NB) {                                   // biasRs
        int l = idx / 256, m = idx % 256;
        float v = 0.f;
        if (l < NLAY - 1)            v = rs_b[idx];
        else if (m < NC)             v = rs_b_last[m];
        biasRs[idx] = v;
    }
}

// ---------------------------------------------------------------------------
// Start conv (4 -> 128), mel transpose to [b][t][80] bf16, zero skip.
// x kept as f32 [b][t][128] (residual state) and bf16 copy for WMMA.
// ---------------------------------------------------------------------------
__global__ void prep_inputs(const float* __restrict__ wavs,
                            const float* __restrict__ mels,
                            const float* __restrict__ start_w,
                            const float* __restrict__ start_b,
                            float* __restrict__ xf,
                            unsigned short* __restrict__ xb0,
                            unsigned short* __restrict__ mel_t,
                            float* __restrict__ skip)
{
    int idx = blockIdx.x * blockDim.x + threadIdx.x;   // over B*T
    if (idx >= BATCH * TLEN) return;
    int b = idx / TLEN, t = idx % TLEN;

    for (int e = 0; e < N_EMB; ++e)
        mel_t[(size_t)idx * N_EMB + e] =
            f2bf(mels[((size_t)b * N_EMB + e) * TLEN + t]);

    float w0 = wavs[((size_t)b * N_IN + 0) * TLEN + t];
    float w1 = wavs[((size_t)b * N_IN + 1) * TLEN + t];
    float w2 = wavs[((size_t)b * N_IN + 2) * TLEN + t];
    float w3 = wavs[((size_t)b * N_IN + 3) * TLEN + t];

    size_t base = (size_t)idx * NC;
    for (int c = 0; c < NC; ++c) {
        float x = start_b[c] + start_w[c * 4 + 0] * w0 + start_w[c * 4 + 1] * w1
                             + start_w[c * 4 + 2] * w2 + start_w[c * 4 + 3] * w3;
        xf[base + c]   = x;
        xb0[base + c]  = f2bf(x);
        skip[base + c] = 0.f;
    }
}

// ---------------------------------------------------------------------------
// Fused WaveNet layer:
//   GEMM1: in_act(256xTT) = Wcat(256x480) @ feat(480xTT)   [bf16 WMMA, f32 acc]
//   gate : acts = tanh(lo+b) * sigmoid(hi+b)               [cross-wave via LDS]
//   GEMM2: rs(256xTT) = Wrs(256x128) @ acts(128xTT)
//   x += rs[:128] (f32 in place, bf16 copy out), skip += rs[128:]
// ---------------------------------------------------------------------------
__global__ __launch_bounds__(256)
void layer_kernel(const unsigned short* __restrict__ xb_in,
                  unsigned short* __restrict__ xb_out,
                  float* __restrict__ xf,
                  float* __restrict__ skip,
                  const unsigned short* __restrict__ mel_t,
                  const unsigned short* __restrict__ Wcat,
                  const unsigned short* __restrict__ Wrs,
                  const float* __restrict__ biasCat,
                  const float* __restrict__ biasRs,
                  int d, int is_last)
{
    __shared__ unsigned short featB[TT * KPAD];   // 62464 B : feat tile [n][k]
    __shared__ float          sig[NC * TT];       // 32768 B : sigmoid half
    __shared__ unsigned short actsB[TT * 136];    // 17408 B : acts tile [n][c]

    const int tid  = threadIdx.x;
    const int wave = tid >> 5;
    const int lane = tid & 31;
    const int hi   = lane >> 4;
    const int ln   = lane & 15;
    const int b    = blockIdx.y;
    const int t0   = blockIdx.x * TT;
    const int m0   = wave * 32;                   // rows m0..m0+31

    // ---- stage feat tile: 4 threads per time-row, one 128/104-elem segment --
    {
        int n   = tid >> 2;
        int seg = tid & 3;
        int t   = t0 + n;
        uint4* dst = (uint4*)&featB[n * KPAD + seg * NC];
        uint4  z;  z.x = z.y = z.z = z.w = 0u;
        if (seg < 3) {
            int tt = t + (seg - 1) * d;
            if (t < TLEN && tt >= 0 && tt < TLEN) {
                const uint4* src = (const uint4*)(xb_in + ((size_t)b * TLEN + tt) * NC);
#pragma unroll
                for (int i = 0; i < 16; ++i) dst[i] = src[i];
            } else {
#pragma unroll
                for (int i = 0; i < 16; ++i) dst[i] = z;
            }
        } else {                                   // mel (80) + zero pad (24)
            if (t < TLEN) {
                const uint4* src = (const uint4*)(mel_t + ((size_t)b * TLEN + t) * N_EMB);
#pragma unroll
                for (int i = 0; i < 10; ++i) dst[i] = src[i];
            } else {
#pragma unroll
                for (int i = 0; i < 10; ++i) dst[i] = z;
            }
#pragma unroll
            for (int i = 10; i < 13; ++i) dst[i] = z;
        }
    }
    __syncthreads();

    // ---- GEMM1: C[mt][nt] += A(Wcat) x B(feat) --------------------------------
    v8f C[2][4] = {};
    for (int kb = 0; kb < KTOT / 32; ++kb) {
        Frag16 A0, A1;
        {   // A layout: lane = row, e<8 -> K=kb*32+hi*8+e ; e>=8 -> +16
            const unsigned short* a0 = Wcat + (size_t)(m0 + ln) * KTOT + kb * 32 + hi * 8;
            const unsigned short* a1 = Wcat + (size_t)(m0 + 16 + ln) * KTOT + kb * 32 + hi * 8;
            A0.q[0] = *(const uint4*)(a0);      A0.q[1] = *(const uint4*)(a0 + 16);
            A1.q[0] = *(const uint4*)(a1);      A1.q[1] = *(const uint4*)(a1 + 16);
        }
#pragma unroll
        for (int nt = 0; nt < 4; ++nt) {
            Frag16 Bv;                           // lane = col, e<8 -> K=kb*32+hi*8+e ; e>=8 -> +16
            const unsigned short* bp = &featB[(nt * 16 + ln) * KPAD + kb * 32 + hi * 8];
            Bv.q[0] = *(const uint4*)(bp);
            Bv.q[1] = *(const uint4*)(bp + 16);
            C[0][nt] = __builtin_amdgcn_wmma_f32_16x16x32_bf16(
                false, A0.v, false, Bv.v, (short)0, C[0][nt], false, false);
            C[1][nt] = __builtin_amdgcn_wmma_f32_16x16x32_bf16(
                false, A1.v, false, Bv.v, (short)0, C[1][nt], false, false);
        }
    }

    // ---- gating: sigmoid half -> LDS, then tanh*sig -> bf16 acts tile --------
    if (m0 >= NC) {
#pragma unroll
        for (int mt = 0; mt < 2; ++mt)
#pragma unroll
            for (int nt = 0; nt < 4; ++nt)
#pragma unroll
                for (int v = 0; v < 8; ++v) {
                    int mg = m0 + mt * 16 + 8 * hi + v;        // 128..255
                    int n  = nt * 16 + ln;
                    float x = C[mt][nt][v] + biasCat[mg];
                    sig[(mg - NC) * TT + n] = 1.f / (1.f + __expf(-x));
                }
    }
    __syncthreads();
    if (m0 < NC) {
#pragma unroll
        for (int mt = 0; mt < 2; ++mt)
#pragma unroll
            for (int nt = 0; nt < 4; ++nt)
#pragma unroll
                for (int v = 0; v < 8; ++v) {
                    int mg = m0 + mt * 16 + 8 * hi + v;        // 0..127
                    int n  = nt * 16 + ln;
                    float x = C[mt][nt][v] + biasCat[mg];
                    float a = tanhf(x) * sig[mg * TT + n];
                    actsB[n * 136 + mg] = f2bf(a);
                }
    }
    __syncthreads();

    // ---- GEMM2: rs = Wrs(256x128) @ acts(128xTT) ------------------------------
    v8f D[2][4] = {};
    for (int kb = 0; kb < NC / 32; ++kb) {
        Frag16 A0, A1;
        {
            const unsigned short* a0 = Wrs + (size_t)(m0 + ln) * NC + kb * 32 + hi * 8;
            const unsigned short* a1 = Wrs + (size_t)(m0 + 16 + ln) * NC + kb * 32 + hi * 8;
            A0.q[0] = *(const uint4*)(a0);      A0.q[1] = *(const uint4*)(a0 + 16);
            A1.q[0] = *(const uint4*)(a1);      A1.q[1] = *(const uint4*)(a1 + 16);
        }
#pragma unroll
        for (int nt = 0; nt < 4; ++nt) {
            Frag16 Bv;
            const unsigned short* bp = &actsB[(nt * 16 + ln) * 136 + kb * 32 + hi * 8];
            Bv.q[0] = *(const uint4*)(bp);
            Bv.q[1] = *(const uint4*)(bp + 16);
            D[0][nt] = __builtin_amdgcn_wmma_f32_16x16x32_bf16(
                false, A0.v, false, Bv.v, (short)0, D[0][nt], false, false);
            D[1][nt] = __builtin_amdgcn_wmma_f32_16x16x32_bf16(
                false, A1.v, false, Bv.v, (short)0, D[1][nt], false, false);
        }
    }

    // ---- epilogue: residual + skip -------------------------------------------
#pragma unroll
    for (int mt = 0; mt < 2; ++mt)
#pragma unroll
        for (int nt = 0; nt < 4; ++nt) {
            int n = nt * 16 + ln;
            int t = t0 + n;
            if (t >= TLEN) continue;
            size_t base = ((size_t)b * TLEN + t) * NC;
#pragma unroll
            for (int v = 0; v < 8; ++v) {
                int mg = m0 + mt * 16 + 8 * hi + v;
                float val = D[mt][nt][v] + biasRs[mg];
                if (!is_last) {
                    if (mg < NC) {
                        float xn = xf[base + mg] + val;
                        xf[base + mg]     = xn;
                        xb_out[base + mg] = f2bf(xn);
                    } else {
                        skip[base + (mg - NC)] += val;
                    }
                } else if (mg < NC) {
                    skip[base + mg] += val;
                }
            }
        }
}

// ---------------------------------------------------------------------------
// End conv (128 -> 8), f32 output in reference (B, 8, T) layout.
// ---------------------------------------------------------------------------
__global__ void end_kernel(const float* __restrict__ skip,
                           const float* __restrict__ end_w,
                           const float* __restrict__ end_b,
                           float* __restrict__ out)
{
    int idx = blockIdx.x * blockDim.x + threadIdx.x;
    if (idx >= BATCH * TLEN) return;
    int b = idx / TLEN, t = idx % TLEN;
    float acc[8];
#pragma unroll
    for (int o = 0; o < 8; ++o) acc[o] = end_b[o];
    const float* s = skip + (size_t)idx * NC;
    for (int c = 0; c < NC; ++c) {
        float sv = s[c];
#pragma unroll
        for (int o = 0; o < 8; ++o) acc[o] += end_w[o * NC + c] * sv;
    }
#pragma unroll
    for (int o = 0; o < 8; ++o)
        out[((size_t)b * 8 + o) * TLEN + t] = acc[o];
}

// ---------------------------------------------------------------------------
extern "C" void kernel_launch(void* const* d_in, const int* in_sizes, int n_in,
                              void* d_out, int out_size, void* d_ws, size_t ws_size,
                              hipStream_t stream)
{
    const float* wavs      = (const float*)d_in[0];
    const float* mels      = (const float*)d_in[1];
    const float* start_w   = (const float*)d_in[2];
    const float* start_b   = (const float*)d_in[3];
    const float* in_w      = (const float*)d_in[4];
    const float* in_b      = (const float*)d_in[5];
    const float* cond_w    = (const float*)d_in[6];
    const float* cond_b    = (const float*)d_in[7];
    const float* rs_w      = (const float*)d_in[8];
    const float* rs_b      = (const float*)d_in[9];
    const float* rs_w_last = (const float*)d_in[10];
    const float* rs_b_last = (const float*)d_in[11];
    const float* end_w     = (const float*)d_in[12];
    const float* end_b     = (const float*)d_in[13];
    float* out = (float*)d_out;

    // workspace carve-up (256B aligned)
    size_t off = 0;
    char* ws = (char*)d_ws;
    auto carve = [&](size_t bytes) -> void* {
        void* p = ws + off;
        off += (bytes + 255) & ~(size_t)255;
        return p;
    };
    const size_t nBT = (size_t)BATCH * TLEN;
    unsigned short* mel_t = (unsigned short*)carve(nBT * N_EMB * 2);
    float*          xf    = (float*)         carve(nBT * NC * 4);
    unsigned short* xb0   = (unsigned short*)carve(nBT * NC * 2);
    unsigned short* xb1   = (unsigned short*)carve(nBT * NC * 2);
    float*          skipb = (float*)         carve(nBT * NC * 4);
    unsigned short* Wcat  = (unsigned short*)carve((size_t)NLAY * 256 * KTOT * 2);
    unsigned short* Wrs   = (unsigned short*)carve((size_t)NLAY * 256 * NC * 2);
    float*          bCat  = (float*)         carve((size_t)NLAY * 256 * 4);
    float*          bRs   = (float*)         carve((size_t)NLAY * 256 * 4);
    (void)ws_size; (void)in_sizes; (void)n_in; (void)out_size;

    {   // pack weights / biases
        int total = NLAY * 256 * KTOT + NLAY * 256 * NC + 2 * NLAY * 256;
        prep_weights<<<(total + 255) / 256, 256, 0, stream>>>(
            in_w, in_b, cond_w, cond_b, rs_w, rs_b, rs_w_last, rs_b_last,
            Wcat, Wrs, bCat, bRs);
    }
    {   // start conv + mel transpose + skip zero
        int total = BATCH * TLEN;
        prep_inputs<<<(total + 255) / 256, 256, 0, stream>>>(
            wavs, mels, start_w, start_b, xf, xb0, mel_t, skipb);
    }

    unsigned short* bufs[2] = { xb0, xb1 };
    dim3 grid((TLEN + TT - 1) / TT, BATCH);
    for (int l = 0; l < NLAY; ++l) {
        layer_kernel<<<grid, 256, 0, stream>>>(
            bufs[l & 1], bufs[(l + 1) & 1], xf, skipb, mel_t,
            Wcat + (size_t)l * 256 * KTOT,
            Wrs  + (size_t)l * 256 * NC,
            bCat + (size_t)l * 256,
            bRs  + (size_t)l * 256,
            1 << l, (l == NLAY - 1) ? 1 : 0);
    }

    {   // end conv
        int total = BATCH * TLEN;
        end_kernel<<<(total + 255) / 256, 256, 0, stream>>>(
            skipb, end_w, end_b, out);
    }
}
